// ProtoNetManifold_34385508172447
// MI455X (gfx1250) — matrix operand order repaired and checked by reading, not compiled, not run on hardware
//
#include <hip/hip_runtime.h>
#include <hip/hip_bf16.h>
#include <math.h>

// ---------------- problem constants ----------------
#define T_TOT   8192      // 4096 support pts + 4096 query pts
#define DIMF    320
#define KNN_K   20
#define CAP     128       // adjacency capacity per row (kNN union)
#define NITER   160       // Chebyshev iterations: 0.868^160 ~ 1e-10
#define ALPHA_C 0.99f

typedef _Float16 h16v __attribute__((ext_vector_type(16)));
typedef _Float16 h8v  __attribute__((ext_vector_type(8)));
typedef float    f8v  __attribute__((ext_vector_type(8)));

// ---------------- prototypes + normalized protos ----------------
__global__ __launch_bounds__(320) void proto_kernel(const float* __restrict__ sf,
                                                    const int* __restrict__ sy,
                                                    float* __restrict__ pn) {
  int d = threadIdx.x;                 // feature index 0..319
  __shared__ float protoS[3][DIMF];
  __shared__ float nrmS[3];
  float fgf[2], bgf[2];
  for (int w = 0; w < 2; ++w) {
    float sfg = 0.f, sbg = 0.f, cfg = 0.f, cbg = 0.f;
    const float* base = sf + ((size_t)w * DIMF + d) * 2048;
    const int* yb = sy + w * 2048;
    for (int p = 0; p < 2048; ++p) {
      float v = base[p];
      float m = (float)yb[p];
      sfg += v * m; sbg += v * (1.f - m);
      cfg += m;     cbg += 1.f - m;
    }
    fgf[w] = sfg / (cfg + 1e-5f);
    bgf[w] = sbg / (cbg + 1e-5f);
  }
  protoS[0][d] = 0.5f * (bgf[0] + bgf[1]);   // bg proto = mean over ways (K_SHOT=1)
  protoS[1][d] = fgf[0];
  protoS[2][d] = fgf[1];
  __syncthreads();
  if (d < 3) {
    float s = 0.f;
    for (int i = 0; i < DIMF; ++i) { float v = protoS[d][i]; s += v * v; }
    nrmS[d] = fmaxf(sqrtf(s), 1e-8f);
  }
  __syncthreads();
  for (int c = 0; c < 3; ++c) pn[c * DIMF + d] = protoS[c][d] / nrmS[c];
}

// ---------------- transpose+scale into f16 emb [T][320] ----------------
__global__ __launch_bounds__(256) void prep_emb_kernel(const float* __restrict__ sf,
                                                       const float* __restrict__ qf,
                                                       _Float16* __restrict__ embH,
                                                       float scale) {
  const int HALF = 2 * DIMF * 2048;
  int tid = blockIdx.x * 256 + threadIdx.x;
  if (tid < HALF) {
    int w = tid / (DIMF * 2048);
    int rem = tid % (DIMF * 2048);
    int d = rem / 2048, p = rem % 2048;
    float v = sf[tid] * scale;
    embH[(size_t)(w * 2048 + p) * DIMF + d] = (_Float16)v;
  } else if (tid < 2 * HALF) {
    int t2 = tid - HALF;
    int n = t2 / (DIMF * 2048);
    int rem = t2 % (DIMF * 2048);
    int d = rem / 2048, p = rem % 2048;
    float v = qf[t2] * scale;
    embH[(size_t)(4096 + n * 2048 + p) * DIMF + d] = (_Float16)v;
  }
}

// ---------------- per-row squared norms ----------------
__global__ __launch_bounds__(32) void sq_kernel(const _Float16* __restrict__ embH,
                                                float* __restrict__ sq) {
  int row = blockIdx.x, lane = threadIdx.x;
  const _Float16* p = embH + (size_t)row * DIMF;
  float s = 0.f;
  for (int i = lane; i < DIMF; i += 32) { float v = (float)p[i]; s += v * v; }
  for (int off = 16; off > 0; off >>= 1) s += __shfl_down(s, off);
  if (lane == 0) sq[row] = s;
}

// ---------------- init: y one-hot, x0=0, d=0, loss accumulators ----------------
__global__ __launch_bounds__(256) void init_kernel(const int* __restrict__ sy,
                                                   float* __restrict__ y,
                                                   float* __restrict__ x0,
                                                   float* __restrict__ db,
                                                   float* __restrict__ acc2) {
  int i = blockIdx.x * 256 + threadIdx.x;
  if (i < 2) acc2[i] = 0.f;
  if (i < T_TOT * 3) {
    int row = i / 3, c = i % 3;
    float yv = 0.f;
    if (row < 4096) yv = (sy[row] == c) ? 1.f : 0.f;
    y[i] = yv; x0[i] = 0.f; db[i] = 0.f;
  }
}

// ---------------- cosine logits + query CE partial ----------------
__global__ __launch_bounds__(256) void qpred_kernel(const float* __restrict__ qf,
                                                    const int* __restrict__ qy,
                                                    const float* __restrict__ pn,
                                                    float* __restrict__ out,
                                                    float* __restrict__ qce) {
  __shared__ float pnS[3 * DIMF];
  for (int i = threadIdx.x; i < 3 * DIMF; i += 256) pnS[i] = pn[i];
  __syncthreads();
  int pid = blockIdx.x * 256 + threadIdx.x;   // 0..4095
  int n = pid >> 11, p = pid & 2047;
  const float* base = qf + (size_t)n * DIMF * 2048 + p;
  float d0 = 0.f, d1 = 0.f, d2 = 0.f, nrm = 0.f;
  for (int d = 0; d < DIMF; ++d) {
    float v = base[(size_t)d * 2048];
    nrm += v * v;
    d0 += v * pnS[d];
    d1 += v * pnS[DIMF + d];
    d2 += v * pnS[2 * DIMF + d];
  }
  float inv = 10.0f / fmaxf(sqrtf(nrm), 1e-8f);   // SCALER folded in
  float l0 = d0 * inv, l1 = d1 * inv, l2 = d2 * inv;
  out[((size_t)n * 3 + 0) * 2048 + p] = l0;
  out[((size_t)n * 3 + 1) * 2048 + p] = l1;
  out[((size_t)n * 3 + 2) * 2048 + p] = l2;
  float mx = fmaxf(l0, fmaxf(l1, l2));
  float lse = mx + logf(expf(l0 - mx) + expf(l1 - mx) + expf(l2 - mx));
  int lab = qy[pid];
  float ll = (lab == 0) ? l0 : ((lab == 1) ? l1 : l2);
  atomicAdd(qce, lse - ll);
}

// ---------------- WMMA Gram + fused bottom-20(d^2) per row ----------------
__global__ __launch_bounds__(32) void gram_topk_kernel(const _Float16* __restrict__ embH,
                                                       const float* __restrict__ sq,
                                                       float* __restrict__ knn_val,
                                                       int* __restrict__ knn_idx) {
  const int lane = threadIdx.x;
  const int hi = lane >> 4, lo = lane & 15;
  const int r0 = blockIdx.x * 16;

  __shared__ float sqAll[T_TOT];      // 32 KB
  __shared__ float tile[16][17];
  __shared__ float kval[16][KNN_K];
  __shared__ int   kidx[16][KNN_K];
  __shared__ float kmax[16];
  __shared__ int   kmaxpos[16];

  for (int i = lane; i < T_TOT; i += 32) sqAll[i] = sq[i];
  if (lane < 16) {
    kmax[lane] = 3.4e38f; kmaxpos[lane] = 0;
    for (int k = 0; k < KNN_K; ++k) { kval[lane][k] = 3.4e38f; kidx[lane][k] = -1; }
  }
  // preload A strip: rows r0..r0+15, all K=320, in ISA 16-bit A layout
  h16v A[10];
  const _Float16* arow = embH + (size_t)(r0 + lo) * DIMF;
#pragma unroll
  for (int c = 0; c < 10; ++c) {
    h8v p0 = *(const h8v*)(arow + 32 * c + 8 * hi);
    h8v p1 = *(const h8v*)(arow + 32 * c + 16 + 8 * hi);
#pragma unroll
    for (int e = 0; e < 8; ++e) { A[c][e] = p0[e]; A[c][8 + e] = p1[e]; }
  }
  float sqRowR[8];
#pragma unroll
  for (int r = 0; r < 8; ++r) sqRowR[r] = sq[r0 + r + 8 * hi];
  __syncthreads();

  const float invDim = 1.0f / (float)DIMF;

  for (int ct = 0; ct < T_TOT / 16; ++ct) {
    const int c0 = ct * 16;
    const _Float16* brow = embH + (size_t)(c0 + lo) * DIMF;
    f8v acc = {};
#pragma unroll
    for (int c = 0; c < 10; ++c) {
      h8v q0 = *(const h8v*)(brow + 32 * c + 8 * hi);
      h8v q1 = *(const h8v*)(brow + 32 * c + 16 + 8 * hi);
      h16v B;
#pragma unroll
      for (int e = 0; e < 8; ++e) { B[e] = q0[e]; B[8 + e] = q1[e]; }
      acc = __builtin_amdgcn_wmma_f32_16x16x32_f16(false, A[c], false, B,
                                                   (short)0, acc, false, false);
    }
    float sqc = sqAll[c0 + lo];
    float d2v[8];
    int cand = 0;
#pragma unroll
    for (int r = 0; r < 8; ++r) {
      int m = r + 8 * hi;
      float d2 = (sqRowR[r] + sqc - 2.0f * acc[r]) * invDim;
      d2v[r] = d2;
      cand |= (d2 < kmax[m]) ? 1 : 0;
    }
    if (__any(cand)) {
#pragma unroll
      for (int r = 0; r < 8; ++r) tile[r + 8 * hi][lo] = d2v[r];
      __syncthreads();
      if (lane < 16) {
        int m = lane;
        float th = kmax[m];
        for (int n = 0; n < 16; ++n) {
          float v = tile[m][n];
          if (v < th) {
            int pos = kmaxpos[m];
            kval[m][pos] = v; kidx[m][pos] = c0 + n;
            float mx = kval[m][0]; int mp = 0;
            for (int k = 1; k < KNN_K; ++k) {
              float kv = kval[m][k];
              if (kv > mx) { mx = kv; mp = k; }
            }
            kmax[m] = mx; kmaxpos[m] = mp; th = mx;
          }
        }
      }
      __syncthreads();
    }
  }
  if (lane < 16) {
    int row = r0 + lane;
    for (int k = 0; k < KNN_K; ++k) {
      knn_val[row * KNN_K + k] = kval[lane][k];   // stores d^2
      knn_idx[row * KNN_K + k] = kidx[lane][k];
    }
  }
}

// ---------------- sparse symmetrized graph build ----------------
__global__ __launch_bounds__(256) void edgesA_kernel(const float* __restrict__ knn_val,
                                                     const int* __restrict__ knn_idx,
                                                     int* __restrict__ adj_idx,
                                                     float* __restrict__ adj_val,
                                                     int* __restrict__ cnt) {
  int row = blockIdx.x * 256 + threadIdx.x;
  if (row >= T_TOT) return;
  cnt[row] = KNN_K;
  for (int k = 0; k < KNN_K; ++k) {
    adj_idx[(size_t)row * CAP + k] = knn_idx[row * KNN_K + k];
    adj_val[(size_t)row * CAP + k] = expf(-0.5f * knn_val[row * KNN_K + k]);
  }
}

__global__ __launch_bounds__(256) void edgesB_kernel(const float* __restrict__ knn_val,
                                                     const int* __restrict__ knn_idx,
                                                     int* __restrict__ adj_idx,
                                                     float* __restrict__ adj_val,
                                                     int* __restrict__ cnt) {
  int e = blockIdx.x * 256 + threadIdx.x;
  if (e >= T_TOT * KNN_K) return;
  int i = e / KNN_K;
  int j = knn_idx[e];
  bool found = false;
  for (int kk = 0; kk < KNN_K; ++kk)
    if (knn_idx[j * KNN_K + kk] == i) { found = true; break; }
  if (!found) {                    // reverse-only edge: append W[j][i] = W[i][j]
    int pos = atomicAdd(&cnt[j], 1);
    if (pos < CAP) {
      adj_idx[(size_t)j * CAP + pos] = i;
      adj_val[(size_t)j * CAP + pos] = expf(-0.5f * knn_val[e]);
    }
  }
}

__global__ __launch_bounds__(256) void edgesC_kernel(const float* __restrict__ adj_val,
                                                     const int* __restrict__ cnt,
                                                     float* __restrict__ dsi) {
  int row = blockIdx.x * 256 + threadIdx.x;
  if (row >= T_TOT) return;
  int nk = cnt[row]; if (nk > CAP) nk = CAP;
  float D = 0.f;
  for (int k = 0; k < nk; ++k) D += adj_val[(size_t)row * CAP + k];
  dsi[row] = sqrtf(1.0f / (D + 2.2204460492503131e-16f));
}

__global__ __launch_bounds__(256) void edgesD_kernel(float* __restrict__ adj_val,
                                                     const int* __restrict__ adj_idx,
                                                     const int* __restrict__ cnt,
                                                     const float* __restrict__ dsi) {
  int row = blockIdx.x * 256 + threadIdx.x;
  if (row >= T_TOT) return;
  int nk = cnt[row]; if (nk > CAP) nk = CAP;
  float s = ALPHA_C * dsi[row];               // fold alpha into matrix: m_ij = a*dsi_i*W_ij*dsi_j
  for (int k = 0; k < nk; ++k)
    adj_val[(size_t)row * CAP + k] *= s * dsi[adj_idx[(size_t)row * CAP + k]];
}

// ---------------- one Chebyshev iteration (SpMV + update, ping-pong x) ----------------
__global__ __launch_bounds__(256) void cheb_kernel(const float* __restrict__ xin,
                                                   float* __restrict__ xout,
                                                   float* __restrict__ db,
                                                   const float* __restrict__ y,
                                                   const int* __restrict__ adj_idx,
                                                   const float* __restrict__ adj_val,
                                                   const int* __restrict__ cnt,
                                                   float c1, float c2) {
  int row = blockIdx.x * 256 + threadIdx.x;
  if (row >= T_TOT) return;
  int nk = cnt[row]; if (nk > CAP) nk = CAP;
  const int* ai = adj_idx + (size_t)row * CAP;
  const float* av = adj_val + (size_t)row * CAP;
  float a0 = 0.f, a1 = 0.f, a2 = 0.f;
  for (int k = 0; k < nk; ++k) {
    int j = ai[k]; float m = av[k];
    a0 += m * xin[j * 3 + 0];
    a1 += m * xin[j * 3 + 1];
    a2 += m * xin[j * 3 + 2];
  }
  // r = y - B x = y - x + (alpha S) x ;  d = c1*d + c2*r ;  x += d
#pragma unroll
  for (int c = 0; c < 3; ++c) {
    float ac = (c == 0) ? a0 : (c == 1) ? a1 : a2;
    float xi = xin[row * 3 + c];
    float r = y[row * 3 + c] - xi + ac;
    float dv = c1 * db[row * 3 + c] + c2 * r;
    db[row * 3 + c] = dv;
    xout[row * 3 + c] = xi + dv;
  }
}

// ---------------- manifold CE partial ----------------
__global__ __launch_bounds__(256) void mce_kernel(const float* __restrict__ x,
                                                  const int* __restrict__ sy,
                                                  const int* __restrict__ qy,
                                                  float* __restrict__ mce) {
  int row = blockIdx.x * 256 + threadIdx.x;
  if (row >= T_TOT) return;
  float l0 = x[row * 3 + 0], l1 = x[row * 3 + 1], l2 = x[row * 3 + 2];
  int lab = (row < 4096) ? sy[row] : qy[row - 4096];
  float mx = fmaxf(l0, fmaxf(l1, l2));
  float lse = mx + logf(expf(l0 - mx) + expf(l1 - mx) + expf(l2 - mx));
  float ll = (lab == 0) ? l0 : ((lab == 1) ? l1 : l2);
  atomicAdd(mce, lse - ll);
}

__global__ void fin_kernel(const float* __restrict__ acc2, float* __restrict__ out) {
  out[12288] = acc2[0] * (1.0f / 4096.0f) + 0.5f * (acc2[1] * (1.0f / 8192.0f));
}

// ---------------- launch ----------------
extern "C" void kernel_launch(void* const* d_in, const int* in_sizes, int n_in,
                              void* d_out, int out_size, void* d_ws, size_t ws_size,
                              hipStream_t stream) {
  (void)in_sizes; (void)n_in; (void)out_size; (void)ws_size;
  const float* sf = (const float*)d_in[0];
  const float* qf = (const float*)d_in[1];
  const int*   sy = (const int*)d_in[2];
  const int*   qy = (const int*)d_in[3];
  float* out = (float*)d_out;

  char* base = (char*)d_ws;
  size_t o = 0;
  auto alloc = [&](size_t bytes) -> char* {
    char* r = base + o;
    o = (o + bytes + 255) & ~(size_t)255;
    return r;
  };
  float*     pn      = (float*)alloc(3 * DIMF * 4);
  _Float16*  embH    = (_Float16*)alloc((size_t)T_TOT * DIMF * 2);
  float*     sq      = (float*)alloc(T_TOT * 4);
  float*     knn_val = (float*)alloc((size_t)T_TOT * KNN_K * 4);
  int*       knn_idx = (int*)alloc((size_t)T_TOT * KNN_K * 4);
  int*       adj_idx = (int*)alloc((size_t)T_TOT * CAP * 4);
  float*     adj_val = (float*)alloc((size_t)T_TOT * CAP * 4);
  int*       cnt     = (int*)alloc(T_TOT * 4);
  float*     dsi     = (float*)alloc(T_TOT * 4);
  float*     yb      = (float*)alloc((size_t)T_TOT * 3 * 4);
  float*     xA      = (float*)alloc((size_t)T_TOT * 3 * 4);
  float*     xB      = (float*)alloc((size_t)T_TOT * 3 * 4);
  float*     db      = (float*)alloc((size_t)T_TOT * 3 * 4);
  float*     acc2    = (float*)alloc(2 * 4);

  const float scale = (float)(1.0 / (0.25 + 2.220446049250313e-16));

  proto_kernel<<<1, 320, 0, stream>>>(sf, sy, pn);
  prep_emb_kernel<<<(2 * 2 * DIMF * 2048 + 255) / 256, 256, 0, stream>>>(sf, qf, embH, scale);
  sq_kernel<<<T_TOT, 32, 0, stream>>>(embH, sq);
  init_kernel<<<(T_TOT * 3 + 255) / 256, 256, 0, stream>>>(sy, yb, xA, db, acc2);
  qpred_kernel<<<16, 256, 0, stream>>>(qf, qy, pn, out, acc2);
  gram_topk_kernel<<<T_TOT / 16, 32, 0, stream>>>(embH, sq, knn_val, knn_idx);
  edgesA_kernel<<<T_TOT / 256, 256, 0, stream>>>(knn_val, knn_idx, adj_idx, adj_val, cnt);
  edgesB_kernel<<<(T_TOT * KNN_K + 255) / 256, 256, 0, stream>>>(knn_val, knn_idx, adj_idx, adj_val, cnt);
  edgesC_kernel<<<T_TOT / 256, 256, 0, stream>>>(adj_val, cnt, dsi);
  edgesD_kernel<<<T_TOT / 256, 256, 0, stream>>>(adj_val, adj_idx, cnt, dsi);

  // Chebyshev on B = I - alpha*S, spectrum in [1-alpha, 1+alpha]
  double theta = 1.0, delta = 0.99;
  double sigma = theta / delta;
  double rho_old = 1.0 / sigma;
  float* xin = xA; float* xout = xB;
  for (int k = 0; k < NITER; ++k) {
    float c1, c2;
    if (k == 0) { c1 = 0.f; c2 = (float)(1.0 / theta); }
    else {
      double rho_new = 1.0 / (2.0 * sigma - rho_old);
      c1 = (float)(rho_new * rho_old);
      c2 = (float)(2.0 * rho_new / delta);
      rho_old = rho_new;
    }
    cheb_kernel<<<T_TOT / 256, 256, 0, stream>>>(xin, xout, db, yb, adj_idx, adj_val, cnt, c1, c2);
    float* t = xin; xin = xout; xout = t;
  }
  mce_kernel<<<T_TOT / 256, 256, 0, stream>>>(xin, sy, qy, acc2 + 1);
  fin_kernel<<<1, 1, 0, stream>>>(acc2, out);
}
